// SimpleFontain_36773509988481
// MI455X (gfx1250) — matrix operand order repaired and checked by reading, not compiled, not run on hardware
//
#include <hip/hip_runtime.h>
#include <hip/hip_bf16.h>

typedef __attribute__((ext_vector_type(16))) __bf16 v16bf;
typedef __attribute__((ext_vector_type(8)))  float  v8f;
typedef __attribute__((ext_vector_type(4)))  int    v4i;

#define T_STEPS 256
#define BATCH   128
#define VOCN    256
#define IN_DIM  258
#define IN_PAD  288
#define EMBN    512
#define NH      1024
#define HH      2048
#define WINN    32
#define OUT_EMB 1024
#define L0IN    3584   // EMB + H + N

#if __has_builtin(__builtin_amdgcn_global_load_async_to_lds_b128)
#define ASYNC_LDS 1
typedef __attribute__((address_space(1))) v4i* gas_p;
typedef __attribute__((address_space(3))) v4i* las_p;
#endif

#if __has_builtin(__builtin_amdgcn_s_wait_asynccnt)
#define WAIT_ASYNC(n) __builtin_amdgcn_s_wait_asynccnt(n)
#else
#define WAIT_ASYNC(n) asm volatile("s_wait_asynccnt %0" ::"i"(n))
#endif

// ---------- scalar helpers ----------
__device__ __forceinline__ unsigned short f2bf(float f) {
  unsigned u = __builtin_bit_cast(unsigned, f);
  u += 0x7FFFu + ((u >> 16) & 1u);
  return (unsigned short)(u >> 16);
}
__device__ __forceinline__ float bf2f(unsigned short h) {
  unsigned u = ((unsigned)h) << 16;
  return __builtin_bit_cast(float, u);
}
__device__ __forceinline__ float sigm(float x) { return 1.0f / (1.0f + __expf(-x)); }

union FragU { uint4 q[2]; v16bf v; };

// ---------- weight repack: fp32 row-major [K,N] -> bf16 WMMA B-fragment blocks ----------
__global__ void pack_w(const float* __restrict__ W, unsigned short* __restrict__ o,
                       int K, int N, int Ktiles) {
  int t = blockIdx.x * blockDim.x + threadIdx.x;
  int ntiles = N >> 4;
  int total = Ktiles * ntiles * 32;
  if (t >= total) return;
  int lane = t & 31;
  int rest = t >> 5;
  int nt = rest % ntiles;
  int kt = rest / ntiles;
  int n  = nt * 16 + (lane & 15);
  int kb = kt * 32 + (lane >> 4) * 16;
  __align__(16) unsigned short vals[16];
#pragma unroll
  for (int i = 0; i < 16; ++i) {
    int k = kb + i;
    vals[i] = (k < K) ? f2bf(W[(size_t)k * N + n]) : (unsigned short)0;
  }
  uint4* dst = (uint4*)(o + (size_t)t * 16);
  dst[0] = *(const uint4*)&vals[0];
  dst[1] = *(const uint4*)&vals[8];
}

// ---------- input pad+convert: x[B,258] fp32 -> bf16 [B,288] ----------
__global__ void xpad_k(const float* __restrict__ x, unsigned short* __restrict__ o) {
  int idx = blockIdx.x * blockDim.x + threadIdx.x;
  if (idx >= BATCH * IN_PAD) return;
  int b = idx / IN_PAD, col = idx % IN_PAD;
  float v = (col < IN_DIM) ? x[(size_t)b * IN_DIM + col] : 0.0f;
  o[idx] = f2bf(v);
}

// ---------- WMMA GEMM ----------
// block = 128 threads = 4 waves, each wave owns a 16Mx64N strip.
// grid = (N/64, 2 [M halves], kchunks [split-K]).
// B panel (4KB per K-step) staged in LDS, double-buffered via async-to-LDS.
// K loop manually unrolled x2 so A fragments live in fixed registers (no rotation).
// EPI: 0 = bias->bf16 ; 1 = (+bias)->fp32 (split-K partial if gridDim.z>1) ;
//      2 = sigmoid(acc)*hs->bf16 ; 3 = relu(acc+bias)->bf16
template <int EPI>
__global__ __launch_bounds__(128) void gemm128(
    const unsigned short* __restrict__ A, int lda,
    const unsigned short* __restrict__ Wp, int K, int N,
    const float* __restrict__ bias,
    void* __restrict__ out, int ldo,
    const unsigned short* __restrict__ hs, int ldhs) {
  __shared__ __align__(16) unsigned char sB[2][4096];

  const int lane   = threadIdx.x & 31;
  const int wave   = threadIdx.x >> 5;                 // 0..3
  const int m0     = blockIdx.y * 64 + wave * 16;
  const int n0     = blockIdx.x * 64;
  const int ntiles = N >> 4;
  const int half   = lane >> 4;
  const int lane16 = lane & 15;

  const int Kt  = K >> 5;
  const int cnt = Kt / gridDim.z;
  const int ktb = blockIdx.z * cnt;

  const int am   = m0 + lane16;
  const int koff = half * 16;
  const int nt0  = n0 >> 4;

  auto stageB = [&](int kt, int bufid) {
    char* g = (char*)(Wp + (((size_t)kt * ntiles + nt0) << 9));
    char* l = (char*)&sB[bufid][0];
    int o = threadIdx.x * 32;
#ifdef ASYNC_LDS
    __builtin_amdgcn_global_load_async_to_lds_b128(
        (gas_p)(uintptr_t)(g + o),
        (las_p)(unsigned)(uintptr_t)(l + o), 0, 0);
    __builtin_amdgcn_global_load_async_to_lds_b128(
        (gas_p)(uintptr_t)(g + o + 16),
        (las_p)(unsigned)(uintptr_t)(l + o + 16), 0, 0);
#else
    *(uint4*)(l + o)      = *(const uint4*)(g + o);
    *(uint4*)(l + o + 16) = *(const uint4*)(g + o + 16);
#endif
  };
  auto loadA = [&](int kt) {
    FragU a;
    const unsigned short* ap = A + (size_t)am * lda + (kt * 32 + koff);
    a.q[0] = *(const uint4*)(ap);
    a.q[1] = *(const uint4*)(ap + 8);
    return a;
  };

  v8f acc0 = {}, acc1 = {}, acc2 = {}, acc3 = {};

  // 4 fragment loads from the given LDS buffer + 4 WMMAs with the given A fragment
  auto compute = [&](int bufid, const FragU& a) {
    const unsigned short* lb = (const unsigned short*)&sB[bufid][0];
    const unsigned short* p0 = lb + lane * 16;
    FragU b0, b1, b2, b3;
    b0.q[0] = *(const uint4*)(p0);          b0.q[1] = *(const uint4*)(p0 + 8);
    b1.q[0] = *(const uint4*)(p0 + 512);    b1.q[1] = *(const uint4*)(p0 + 520);
    b2.q[0] = *(const uint4*)(p0 + 1024);   b2.q[1] = *(const uint4*)(p0 + 1032);
    b3.q[0] = *(const uint4*)(p0 + 1536);   b3.q[1] = *(const uint4*)(p0 + 1544);
    acc0 = __builtin_amdgcn_wmma_f32_16x16x32_bf16(false, a.v, false, b0.v, (short)0, acc0, false, false);
    acc1 = __builtin_amdgcn_wmma_f32_16x16x32_bf16(false, a.v, false, b1.v, (short)0, acc1, false, false);
    acc2 = __builtin_amdgcn_wmma_f32_16x16x32_bf16(false, a.v, false, b2.v, (short)0, acc2, false, false);
    acc3 = __builtin_amdgcn_wmma_f32_16x16x32_bf16(false, a.v, false, b3.v, (short)0, acc3, false, false);
  };

  FragU aA = loadA(ktb), aB;
  stageB(ktb, 0);

  for (int i = 0; i < cnt; i += 2) {
    // ---- even K-step: compute sB[0] with aA, prefetch sB[1]/aB ----
    const bool m1 = (i + 1 < cnt);
    if (m1) { stageB(ktb + i + 1, 1); aB = loadA(ktb + i + 1); }
#ifdef ASYNC_LDS
    if (m1) { WAIT_ASYNC(2); } else { WAIT_ASYNC(0); }
#endif
    __syncthreads();
    compute(0, aA);
    __syncthreads();
    if (!m1) break;

    // ---- odd K-step: compute sB[1] with aB, prefetch sB[0]/aA ----
    const bool m2 = (i + 2 < cnt);
    if (m2) { stageB(ktb + i + 2, 0); aA = loadA(ktb + i + 2); }
#ifdef ASYNC_LDS
    if (m2) { WAIT_ASYNC(2); } else { WAIT_ASYNC(0); }
#endif
    __syncthreads();
    compute(1, aB);
    __syncthreads();
  }

  // split-K partial target
  if (EPI == 1 && gridDim.z > 1)
    out = (void*)((float*)out + (size_t)blockIdx.z * BATCH * N);

  const int mbase = m0 + half * 8;
#pragma unroll
  for (int j = 0; j < 4; ++j) {
    v8f acc = (j == 0) ? acc0 : (j == 1) ? acc1 : (j == 2) ? acc2 : acc3;
    const int n  = n0 + j * 16 + lane16;
    const float bv = bias ? bias[n] : 0.0f;
#pragma unroll
    for (int r = 0; r < 8; ++r) {
      const int m = mbase + r;
      float v = acc[r] + bv;
      if (EPI == 0) {
        ((unsigned short*)out)[(size_t)m * ldo + n] = f2bf(v);
      } else if (EPI == 1) {
        ((float*)out)[(size_t)m * ldo + n] = v;
      } else if (EPI == 2) {
        float g = sigm(v);
        float h = bf2f(hs[(size_t)m * ldhs + n]);
        ((unsigned short*)out)[(size_t)m * ldo + n] = f2bf(g * h);
      } else {
        ((unsigned short*)out)[(size_t)m * ldo + n] = f2bf(v > 0.0f ? v : 0.0f);
      }
    }
  }
}

// ---------- LSTM pointwise: z = za + zb + bias ; update c ; scatter h (bf16) ----------
__global__ void lstm_point(const float* __restrict__ za, const float* __restrict__ zb,
                           const float* __restrict__ bias, float* __restrict__ c,
                           unsigned short* d1, int s1,
                           unsigned short* d2, int s2,
                           unsigned short* d3, int s3) {
  int idx = blockIdx.x * blockDim.x + threadIdx.x;   // BATCH*NH
  if (idx >= BATCH * NH) return;
  int b = idx >> 10, j = idx & (NH - 1);
  const size_t base = (size_t)b * (4 * NH);
  float zf = za[base + j]          + zb[base + j]          + bias[j];
  float zi = za[base + NH + j]     + zb[base + NH + j]     + bias[NH + j];
  float zo = za[base + 2 * NH + j] + zb[base + 2 * NH + j] + bias[2 * NH + j];
  float zm = za[base + 3 * NH + j] + zb[base + 3 * NH + j] + bias[3 * NH + j];
  float f = sigm(zf), i = sigm(zi), o = sigm(zo), m = tanhf(zm);
  float cv = c[idx] * f + m * i;
  c[idx] = cv;
  unsigned short hb = f2bf(tanhf(cv) * o);
  if (d1) d1[(size_t)b * s1 + j] = hb;
  if (d2) d2[(size_t)b * s2 + j] = hb;
  if (d3) d3[(size_t)b * s3 + j] = hb;
}

// ---------- attention over WIN=32 ring buffer (order-invariant softmax) ----------
__global__ __launch_bounds__(256) void attn_k(const unsigned short* __restrict__ buf, int slot,
                                              unsigned short* __restrict__ dst, int ldd) {
  __shared__ float red[256];
  __shared__ float sc[WINN];
  const int b = blockIdx.x, tid = threadIdx.x;
  const unsigned short* hs = buf + ((size_t)slot * BATCH + b) * HH;
  for (int w = 0; w < WINN; ++w) {
    const unsigned short* row = buf + ((size_t)w * BATCH + b) * HH;
    float s = 0.0f;
    for (int h = tid; h < HH; h += 256) s += bf2f(row[h]) * bf2f(hs[h]);
    red[tid] = s;
    __syncthreads();
    for (int st = 128; st > 0; st >>= 1) {
      if (tid < st) red[tid] += red[tid + st];
      __syncthreads();
    }
    if (tid == 0) sc[w] = red[0];
    __syncthreads();
  }
  if (tid == 0) {
    float mx = sc[0];
    for (int w = 1; w < WINN; ++w) mx = fmaxf(mx, sc[w]);
    float sum = 0.0f;
    for (int w = 0; w < WINN; ++w) { sc[w] = __expf(sc[w] - mx); sum += sc[w]; }
    float inv = 1.0f / sum;
    for (int w = 0; w < WINN; ++w) sc[w] *= inv;
  }
  __syncthreads();
  for (int h = tid; h < HH; h += 256) {
    float a = 0.0f;
    for (int w = 0; w < WINN; ++w) a += sc[w] * bf2f(buf[((size_t)w * BATCH + b) * HH + h]);
    dst[(size_t)b * ldd + h] = f2bf(a);
  }
}

// ---------- host orchestration ----------
static inline size_t align256(size_t x) { return (x + 255) & ~(size_t)255; }

extern "C" void kernel_launch(void* const* d_in, const int* in_sizes, int n_in,
                              void* d_out, int out_size, void* d_ws, size_t ws_size,
                              hipStream_t stream) {
  const float* inp   = (const float*)d_in[0];
  const float* emb_W = (const float*)d_in[1];
  const float* emb_b = (const float*)d_in[2];
  const float* M0    = (const float*)d_in[3];
  const float* B0    = (const float*)d_in[4];
  const float* M1    = (const float*)d_in[5];
  const float* B1    = (const float*)d_in[6];
  const float* Wg    = (const float*)d_in[7];
  const float* We    = (const float*)d_in[8];
  const float* be    = (const float*)d_in[9];
  const float* Wo    = (const float*)d_in[10];
  const float* bo    = (const float*)d_in[11];

  char* ws = (char*)d_ws;
  size_t off = 0;
  auto carve = [&](size_t bytes) { char* p = ws + off; off += align256(bytes); return p; };
  unsigned short* M0p  = (unsigned short*)carve((size_t)L0IN * 4096 * 2);
  unsigned short* M1p  = (unsigned short*)carve((size_t)HH * 4096 * 2);
  unsigned short* Wgp  = (unsigned short*)carve((size_t)HH * HH * 2);
  unsigned short* Wep  = (unsigned short*)carve((size_t)HH * OUT_EMB * 2);
  unsigned short* Wop  = (unsigned short*)carve((size_t)OUT_EMB * VOCN * 2);
  unsigned short* embp = (unsigned short*)carve((size_t)IN_PAD * EMBN * 2);
  unsigned short* xp   = (unsigned short*)carve((size_t)BATCH * IN_PAD * 2);
  unsigned short* l0in = (unsigned short*)carve((size_t)BATCH * L0IN * 2);
  unsigned short* l1in = (unsigned short*)carve((size_t)BATCH * HH * 2);
  float*          z0   = (float*)carve((size_t)2 * BATCH * 4096 * 4);   // split-K partials
  float*          z1   = (float*)carve((size_t)2 * BATCH * 4096 * 4);
  float*          c0   = (float*)carve((size_t)BATCH * NH * 4);
  float*          c1   = (float*)carve((size_t)BATCH * NH * 4);
  unsigned short* buf  = (unsigned short*)carve((size_t)WINN * BATCH * HH * 2);
  unsigned short* gated= (unsigned short*)carve((size_t)BATCH * HH * 2);
  unsigned short* act  = (unsigned short*)carve((size_t)BATCH * OUT_EMB * 2);
  (void)ws_size; (void)in_sizes; (void)n_in; (void)out_size;

  (void)hipMemsetAsync(l0in, 0, (size_t)BATCH * L0IN * 2, stream);
  (void)hipMemsetAsync(l1in, 0, (size_t)BATCH * HH * 2, stream);
  (void)hipMemsetAsync(c0,   0, (size_t)BATCH * NH * 4, stream);
  (void)hipMemsetAsync(c1,   0, (size_t)BATCH * NH * 4, stream);
  (void)hipMemsetAsync(buf,  0, (size_t)WINN * BATCH * HH * 2, stream);

  auto pack = [&](const float* W, unsigned short* o, int K, int N) {
    int Ktiles = (K + 31) / 32;
    int total = Ktiles * (N >> 4) * 32;
    pack_w<<<(total + 255) / 256, 256, 0, stream>>>(W, o, K, N, Ktiles);
  };
  pack(M0,    M0p,  L0IN,    4096);
  pack(M1,    M1p,  HH,      4096);
  pack(Wg,    Wgp,  HH,      HH);
  pack(We,    Wep,  HH,      OUT_EMB);
  pack(Wo,    Wop,  OUT_EMB, VOCN);
  pack(emb_W, embp, IN_DIM,  EMBN);

  float* out = (float*)d_out;   // [T, B, VOC] fp32

  for (int t = 0; t < T_STEPS; ++t) {
    const int slot = t % WINN;
    unsigned short* bufslot = buf + (size_t)slot * BATCH * HH;

    // 1) pad+convert x[t]
    xpad_k<<<(BATCH * IN_PAD + 255) / 256, 256, 0, stream>>>(
        inp + (size_t)t * BATCH * IN_DIM, xp);

    // 2) e = x @ emb_W + emb_b -> bf16 into lstm0_in cols [0,512)
    gemm128<0><<<dim3(EMBN / 64, 2, 1), 128, 0, stream>>>(
        xp, IN_PAD, embp, IN_PAD, EMBN, emb_b, l0in, L0IN, nullptr, 0);

    // 3) z0 partials = [e|from_att|h0] @ M0  (split-K x2, bias added in pointwise)
    gemm128<1><<<dim3(4096 / 64, 2, 2), 128, 0, stream>>>(
        l0in, L0IN, M0p, L0IN, 4096, nullptr, z0, 4096, nullptr, 0);

    // 4) LSTM0 pointwise
    lstm_point<<<(BATCH * NH + 255) / 256, 256, 0, stream>>>(
        z0, z0 + (size_t)BATCH * 4096, B0, c0,
        l0in + (EMBN + HH), L0IN, l1in, HH, bufslot, HH);

    // 5) z1 partials = [h0|h1] @ M1  (split-K x2)
    gemm128<1><<<dim3(4096 / 64, 2, 2), 128, 0, stream>>>(
        l1in, HH, M1p, HH, 4096, nullptr, z1, 4096, nullptr, 0);

    // 6) LSTM1 pointwise
    lstm_point<<<(BATCH * NH + 255) / 256, 256, 0, stream>>>(
        z1, z1 + (size_t)BATCH * 4096, B1, c1,
        l1in + NH, HH, bufslot + NH, HH, nullptr, 0);

    // 7) attention; from_att -> lstm0_in cols [512,2560) for next step
    attn_k<<<BATCH, 256, 0, stream>>>(buf, slot, l0in + EMBN, L0IN);

    // 8) gated = sigmoid(hs @ Wg) * hs   (hs == buf[slot])
    gemm128<2><<<dim3(HH / 64, 2, 1), 128, 0, stream>>>(
        bufslot, HH, Wgp, HH, HH, nullptr, gated, HH, bufslot, HH);

    // 9) act = relu(gated @ We + be)
    gemm128<3><<<dim3(OUT_EMB / 64, 2, 1), 128, 0, stream>>>(
        gated, HH, Wep, HH, OUT_EMB, be, act, OUT_EMB, nullptr, 0);

    // 10) logits = act @ Wo + bo -> fp32 d_out[t]
    gemm128<1><<<dim3(VOCN / 64, 2, 1), 128, 0, stream>>>(
        act, OUT_EMB, Wop, OUT_EMB, VOCN, bo,
        out + (size_t)t * BATCH * VOCN, VOCN, nullptr, 0);
  }
}
